// Narxnn_71631464563329
// MI455X (gfx1250) — matrix operand order repaired and checked by reading, not compiled, not run on hardware
//
#include <hip/hip_runtime.h>

#define NT_    512
#define NGRID_ 2048
#define NX_    16
#define HID_   64
#define DLAG_  4
#define TOUT_  (NT_ - DLAG_)                 // 508
#define TILES_ (TOUT_ * NGRID_ / 16)         // 65024
#define NBLK_  508
#define NWAVE_ 8
#define NTHR_  (NWAVE_ * 32)
#define ITERS_ (TILES_ / (NBLK_ * NWAVE_))   // 16 exactly

typedef __attribute__((ext_vector_type(16))) _Float16 v16h;
typedef __attribute__((ext_vector_type(8)))  _Float16 v8h;
typedef __attribute__((ext_vector_type(2)))  _Float16 v2h;
typedef __attribute__((ext_vector_type(2)))  __fp16   v2hf;   // cvt_pkrtz native type
typedef __attribute__((ext_vector_type(8)))  float    v8f;

__device__ __forceinline__ float hw_tanh(float v) {
#if __has_builtin(__builtin_amdgcn_tanhf)
    return __builtin_amdgcn_tanhf(v);     // V_TANH_F32 (gfx1250 trans op)
#else
    return tanhf(v);
#endif
}

// packed f16 relu: single V_PK_MAX_NUM_F16 (no canonicalize pair)
__device__ __forceinline__ v2hf pk_relu(v2hf p) {
    v2hf r;
    __asm__("v_pk_max_num_f16 %0, %1, 0" : "=v"(r) : "v"(p));
    return r;
}

__global__ __launch_bounds__(NTHR_) void narx_zero(float* __restrict__ out) {
    int i = blockIdx.x * NTHR_ + threadIdx.x;
    if (i < DLAG_ * NGRID_) out[i] = 0.0f;
}

__global__ __launch_bounds__(NTHR_) void narx_fwd(
    const float* __restrict__ x,
    const float* __restrict__ W_in,  const float* __restrict__ b_in,
    const float* __restrict__ W_xh,  const float* __restrict__ W_hh,
    const float* __restrict__ b_h,
    const float* __restrict__ W_out, const float* __restrict__ b_out,
    float* __restrict__ out)
{
    // Transposed f16 weights: row = output column n, cols = K.
    // sWinT K slot 16 carries b_in (bias folded via augmented input x[16]=1).
    __shared__ __align__(32) _Float16 sWinT[HID_][32];
    __shared__ __align__(32) _Float16 sWxhT[HID_][HID_];
    __shared__ __align__(32) _Float16 sWhhT[HID_][HID_];
    __shared__ float sb_h[HID_], sWout[HID_];
    // Wave-private staging: C-layout -> A-layout transpose of u and h.
    // Same-wave LDS ops complete in order => no block barrier needed.
    __shared__ __align__(32) _Float16 stgU[NWAVE_][16][HID_];
    __shared__ __align__(32) _Float16 stgH[NWAVE_][16][HID_];

    const int tid = threadIdx.x;
    for (int i = tid; i < HID_ * 32; i += NTHR_) {
        int n = i >> 5, k = i & 31;
        _Float16 v = (_Float16)0.0f;
        if (k < NX_)       v = (_Float16)W_in[k * HID_ + n];
        else if (k == NX_) v = (_Float16)b_in[n];
        sWinT[n][k] = v;
    }
    for (int i = tid; i < HID_ * HID_; i += NTHR_) {
        int n = i >> 6, k = i & 63;
        sWxhT[n][k] = (_Float16)W_xh[k * HID_ + n];
        sWhhT[n][k] = (_Float16)W_hh[k * HID_ + n];
    }
    if (tid < HID_) { sb_h[tid] = b_h[tid]; sWout[tid] = W_out[tid]; }
    __syncthreads();   // weights shared block-wide: the only real barrier

    const int wave = tid >> 5;
    const int lane = tid & 31;
    const int m    = lane & 15;    // A row, or B/C column N
    const int hi   = lane >> 4;    // lane-half select
    const float bo = b_out[0];

    // Hoist all weight B-fragments into registers (loop-invariant for kernel).
    v16h fWin[4], fWxh[4][2], fWhh[4][2];
    #pragma unroll
    for (int nt = 0; nt < 4; ++nt) {
        fWin[nt] = __builtin_shufflevector(*(const v8h*)&sWinT[nt * 16 + m][8 * hi], *(const v8h*)&sWinT[nt * 16 + m][16 + 8 * hi], 0,1,2,3,4,5,6,7,8,9,10,11,12,13,14,15);
        #pragma unroll
        for (int kt = 0; kt < 2; ++kt) {
            fWxh[nt][kt] = __builtin_shufflevector(*(const v8h*)&sWxhT[nt * 16 + m][32 * kt + 8 * hi], *(const v8h*)&sWxhT[nt * 16 + m][32 * kt + 16 + 8 * hi], 0,1,2,3,4,5,6,7,8,9,10,11,12,13,14,15);
            fWhh[nt][kt] = __builtin_shufflevector(*(const v8h*)&sWhhT[nt * 16 + m][32 * kt + 8 * hi], *(const v8h*)&sWhhT[nt * 16 + m][32 * kt + 16 + 8 * hi], 0,1,2,3,4,5,6,7,8,9,10,11,12,13,14,15);
        }
    }
    float bh_s[4], wo_s[4];
    #pragma unroll
    for (int nt = 0; nt < 4; ++nt) {
        bh_s[nt] = sb_h[nt * 16 + m];
        wo_s[nt] = sWout[nt * 16 + m];
    }

    _Float16 (*su)[HID_] = stgU[wave];
    _Float16 (*sh)[HID_] = stgH[wave];

    for (int it = 0; it < ITERS_; ++it) {
        const int tile  = blockIdx.x * NWAVE_ + wave + it * (NBLK_ * NWAVE_);
        const size_t base = (size_t)tile * 16;   // flat row b = t'*NGRID + g
        v8f hacc[4];

        #pragma unroll
        for (int d = 0; d < DLAG_; ++d) {
            // ---- A fragment of X_d: 16 rows x K16, slot K=16 = 1.0 (bias) ----
            const float* xr = x + (base + (size_t)d * NGRID_ + (size_t)m) * NX_ + 8 * hi;
            float4 f0 = *(const float4*)(xr);
            float4 f1 = *(const float4*)(xr + 4);
            v16h ax;
            ax[0] = (_Float16)f0.x; ax[1] = (_Float16)f0.y;
            ax[2] = (_Float16)f0.z; ax[3] = (_Float16)f0.w;
            ax[4] = (_Float16)f1.x; ax[5] = (_Float16)f1.y;
            ax[6] = (_Float16)f1.z; ax[7] = (_Float16)f1.w;
            ax[8] = (hi == 0) ? (_Float16)1.0f : (_Float16)0.0f;  // K=16 bias row
            #pragma unroll
            for (int k = 9; k < 16; ++k) ax[k] = (_Float16)0.0f;

            // ---- GEMM1: U = [X,1] @ [W_in; b_in]  (C = inline 0) ----
            v8f uacc[4];
            #pragma unroll
            for (int nt = 0; nt < 4; ++nt) {
                v8f c = {0.f, 0.f, 0.f, 0.f, 0.f, 0.f, 0.f, 0.f};
                uacc[nt] = __builtin_amdgcn_wmma_f32_16x16x32_f16(
                    false, ax, false, fWin[nt], (short)0, c, false, false);
            }

            // stage relu(u) and previous h as f16 (row-major [m][k]);
            // pkrtz pairs + packed-f16 relu; compiler fence keeps DS order,
            // HW keeps same-wave LDS in-order (wave-private buffers).
            __asm__ volatile("" ::: "memory");
            #pragma unroll
            for (int nt = 0; nt < 4; ++nt) {
                #pragma unroll
                for (int j = 0; j < 8; j += 2) {
                    v2h ru = __builtin_bit_cast(v2h, pk_relu(
                        __builtin_amdgcn_cvt_pkrtz(uacc[nt][j], uacc[nt][j + 1])));
                    su[j     + 8 * hi][nt * 16 + m] = ru[0];
                    su[j + 1 + 8 * hi][nt * 16 + m] = ru[1];
                    if (d > 0) {
                        v2h ph = __builtin_bit_cast(v2h,
                            __builtin_amdgcn_cvt_pkrtz(hacc[nt][j], hacc[nt][j + 1]));
                        sh[j     + 8 * hi][nt * 16 + m] = ph[0];
                        sh[j + 1 + 8 * hi][nt * 16 + m] = ph[1];
                    }
                }
            }
            __asm__ volatile("" ::: "memory");

            // A fragments of U (and H) for K=64 (two 16x32 k-tiles)
            v8h ul0 = *(const v8h*)&su[m][ 0 + 8 * hi];
            v8h uh0 = *(const v8h*)&su[m][16 + 8 * hi];
            v8h ul1 = *(const v8h*)&su[m][32 + 8 * hi];
            v8h uh1 = *(const v8h*)&su[m][48 + 8 * hi];
            v16h au0 = __builtin_shufflevector(ul0, uh0, 0,1,2,3,4,5,6,7,8,9,10,11,12,13,14,15);
            v16h au1 = __builtin_shufflevector(ul1, uh1, 0,1,2,3,4,5,6,7,8,9,10,11,12,13,14,15);
            v16h ah0, ah1;
            if (d > 0) {
                v8h hl0 = *(const v8h*)&sh[m][ 0 + 8 * hi];
                v8h hh0 = *(const v8h*)&sh[m][16 + 8 * hi];
                v8h hl1 = *(const v8h*)&sh[m][32 + 8 * hi];
                v8h hh1 = *(const v8h*)&sh[m][48 + 8 * hi];
                ah0 = __builtin_shufflevector(hl0, hh0, 0,1,2,3,4,5,6,7,8,9,10,11,12,13,14,15);
                ah1 = __builtin_shufflevector(hl1, hh1, 0,1,2,3,4,5,6,7,8,9,10,11,12,13,14,15);
            }

            // ---- GEMM2: h = tanh(U @ W_xh + H @ W_hh + b_h) ----
            #pragma unroll
            for (int nt = 0; nt < 4; ++nt) {
                float bh = bh_s[nt];
                v8f c = {bh, bh, bh, bh, bh, bh, bh, bh};
                c = __builtin_amdgcn_wmma_f32_16x16x32_f16(false, au0, false, fWxh[nt][0], (short)0, c, false, false);
                c = __builtin_amdgcn_wmma_f32_16x16x32_f16(false, au1, false, fWxh[nt][1], (short)0, c, false, false);
                if (d > 0) {
                    c = __builtin_amdgcn_wmma_f32_16x16x32_f16(false, ah0, false, fWhh[nt][0], (short)0, c, false, false);
                    c = __builtin_amdgcn_wmma_f32_16x16x32_f16(false, ah1, false, fWhh[nt][1], (short)0, c, false, false);
                }
                #pragma unroll
                for (int j = 0; j < 8; ++j) c[j] = hw_tanh(c[j]);
                hacc[nt] = c;
            }
        }

        // ---- output: y = h @ W_out + b_out (C layout: row j+8*hi, col nt*16+m) ----
        #pragma unroll
        for (int j = 0; j < 8; ++j) {
            float s = 0.0f;
            #pragma unroll
            for (int nt = 0; nt < 4; ++nt) s += hacc[nt][j] * wo_s[nt];
            #pragma unroll
            for (int off = 1; off < 16; off <<= 1) s += __shfl_xor(s, off, 32);
            if (m == 0)
                out[base + (size_t)DLAG_ * NGRID_ + 8 * hi + j] = s + bo;
        }
    }
}

extern "C" void kernel_launch(void* const* d_in, const int* in_sizes, int n_in,
                              void* d_out, int out_size, void* d_ws, size_t ws_size,
                              hipStream_t stream) {
    const float* x     = (const float*)d_in[0];
    const float* W_in  = (const float*)d_in[1];
    const float* b_in  = (const float*)d_in[2];
    const float* W_xh  = (const float*)d_in[3];
    const float* W_hh  = (const float*)d_in[4];
    const float* b_h   = (const float*)d_in[5];
    const float* W_out = (const float*)d_in[6];
    const float* b_out = (const float*)d_in[7];
    float* out = (float*)d_out;

    narx_zero<<<(DLAG_ * NGRID_ + NTHR_ - 1) / NTHR_, NTHR_, 0, stream>>>(out);
    narx_fwd<<<NBLK_, NTHR_, 0, stream>>>(x, W_in, b_in, W_xh, W_hh, b_h, W_out, b_out, out);
}
